// Attention_19739669693111
// MI455X (gfx1250) — matrix operand order repaired and checked by reading, not compiled, not run on hardware
//
#include <hip/hip_runtime.h>

typedef __attribute__((ext_vector_type(16))) _Float16 v16h;
typedef __attribute__((ext_vector_type(8)))  _Float16 v8h;
typedef __attribute__((ext_vector_type(8)))  float    v8f;

#define NHEAD  32
#define NKV    8
#define DHEAD  128
#define WINDOW 1024
#define SCALE_F 0.08838834764831845f
#define NEG_INF_F (-1e30f)
// ln(10000)/64 : inv_freq[i] = exp(-i * this)
#define LN_BASE_OVER_HALFD 0.14391156360544947f

#define QS_STRIDE 136   // halves; 272B rows: 16B aligned, bank stride 68 dwords
#define KS_STRIDE 136
#define VT_STRIDE 40    // 80B rows
#define PS_STRIDE 40

static __device__ __forceinline__ v16h ld8x2(const _Float16* p0, const _Float16* p1) {
    v8h a = *(const v8h*)p0;
    v8h b = *(const v8h*)p1;
    return __builtin_shufflevector(a, b, 0,1,2,3,4,5,6,7,8,9,10,11,12,13,14,15);
}

static __device__ __forceinline__ v8f wmma16(v16h a, v16h b, v8f c) {
    return __builtin_amdgcn_wmma_f32_16x16x32_f16(
        /*neg_a=*/false, a, /*neg_b=*/false, b,
        /*c_mod=*/(short)0, c, /*reuse_a=*/false, /*reuse_b=*/false);
}

__global__ __launch_bounds__(128)
void attn_swa_sink_kernel(const float* __restrict__ q,
                          const float* __restrict__ k,
                          const float* __restrict__ v,
                          const int*   __restrict__ positions,
                          const float* __restrict__ sinks,
                          float*       __restrict__ out,
                          int S)
{
    __shared__ _Float16 Qs[64 * QS_STRIDE];
    __shared__ _Float16 Ks[32 * KS_STRIDE];
    __shared__ _Float16 Vt[128 * VT_STRIDE];
    __shared__ _Float16 Ps[4 * 16 * PS_STRIDE];

    const int t    = threadIdx.x;
    const int wid  = t >> 5;
    const int lane = t & 31;
    const int ln   = lane & 15;   // column / row-within-16 selector
    const int hi   = lane >> 4;   // half-wave selector
    const int qb   = blockIdx.x * 64;
    const int h    = blockIdx.y;
    const int kvh  = h >> 2;      // H/KV = 4

    // ---- Stage RoPE'd Q tile (64 rows x 128) into LDS as f16 ----
    for (int i = 0; i < 32; ++i) {
        int w   = i * 128 + t;         // 0..4095
        int row = w >> 6;              // 0..63
        int pr  = w & 63;              // rotary pair index
        int qg  = qb + row;
        const float* qp = q + ((size_t)qg * NHEAD + h) * DHEAD;
        float x1 = qp[pr];
        float x2 = qp[pr + 64];
        float ang = (float)positions[qg] * __expf(-(float)pr * LN_BASE_OVER_HALFD);
        float sn, cs;
        __sincosf(ang, &sn, &cs);
        Qs[row * QS_STRIDE + pr]      = (_Float16)(x1 * cs - x2 * sn);
        Qs[row * QS_STRIDE + pr + 64] = (_Float16)(x2 * cs + x1 * sn);
    }
    __syncthreads();

    // ---- Per-wave Q A-fragments (4 chunks of K=32) ----
    // A 16x32 layout: lane row = lane&15; lane<16 holds K {0..7,16..23}, lane>=16 holds {8..15,24..31}
    v16h qf[4];
#pragma unroll
    for (int c = 0; c < 4; ++c) {
        const _Float16* p = &Qs[(wid * 16 + ln) * QS_STRIDE + c * 32 + hi * 8];
        qf[c] = ld8x2(p, p + 16);
    }

    v8f acc[8];
#pragma unroll
    for (int c = 0; c < 8; ++c)
#pragma unroll
        for (int r = 0; r < 8; ++r) acc[c][r] = 0.0f;

    float m_run[8], l_run[8];
    const float sink = sinks[h];
#pragma unroll
    for (int r = 0; r < 8; ++r) { m_run[r] = sink; l_run[r] = 1.0f; }

    int klo = qb - (WINDOW - 1);
    if (klo < 0) klo = 0;
    klo &= ~31;
    const int khi = qb + 63;

    for (int kv0 = klo; kv0 <= khi; kv0 += 32) {
        __syncthreads();
        // ---- Stage RoPE'd K block (32 x 128) ----
        for (int i = 0; i < 16; ++i) {
            int w   = i * 128 + t;
            int row = w >> 6;          // 0..31
            int pr  = w & 63;
            int kg  = kv0 + row;
            float r1 = 0.0f, r2 = 0.0f;
            if (kg < S) {
                const float* kp = k + ((size_t)kg * NKV + kvh) * DHEAD;
                float x1 = kp[pr];
                float x2 = kp[pr + 64];
                float ang = (float)positions[kg] * __expf(-(float)pr * LN_BASE_OVER_HALFD);
                float sn, cs;
                __sincosf(ang, &sn, &cs);
                r1 = x1 * cs - x2 * sn;
                r2 = x2 * cs + x1 * sn;
            }
            Ks[row * KS_STRIDE + pr]      = (_Float16)r1;
            Ks[row * KS_STRIDE + pr + 64] = (_Float16)r2;
        }
        // ---- Stage V block transposed: Vt[d][key] ----
        for (int i = 0; i < 32; ++i) {
            int row = i;       // key within block
            int d   = t;       // t in 0..127
            int kg  = kv0 + row;
            float val = (kg < S) ? v[((size_t)kg * NKV + kvh) * DHEAD + d] : 0.0f;
            Vt[d * VT_STRIDE + row] = (_Float16)val;
        }
        __syncthreads();

        // ---- Scores: S(16q x 32k) as two 16x16 WMMA accumulations over 4 d-chunks ----
        v8f s0, s1;
#pragma unroll
        for (int r = 0; r < 8; ++r) { s0[r] = 0.0f; s1[r] = 0.0f; }
#pragma unroll
        for (int c = 0; c < 4; ++c) {
            // B 32x16: lane col = ln, contraction d = c*32 + {hi*8..+7, 16+hi*8..+7}
            const _Float16* p0 = &Ks[ln * KS_STRIDE + c * 32 + hi * 8];
            const _Float16* p1 = &Ks[(16 + ln) * KS_STRIDE + c * 32 + hi * 8];
            v16h kb0 = ld8x2(p0, p0 + 16);
            v16h kb1 = ld8x2(p1, p1 + 16);
            s0 = wmma16(qf[c], kb0, s0);
            s1 = wmma16(qf[c], kb1, s1);
        }

        // ---- Online softmax (C layout: slot r -> row r + 8*hi, lane%16 -> col) ----
        const int qrow0 = qb + wid * 16 + hi * 8;
        float alpha[8], p0v[8], p1v[8];
#pragma unroll
        for (int r = 0; r < 8; ++r) {
            int qg = qrow0 + r;
            int k0 = kv0 + ln;
            int k1 = k0 + 16;
            float v0 = s0[r] * SCALE_F;
            float v1 = s1[r] * SCALE_F;
            bool a0 = (k0 <= qg) && (qg - k0 < WINDOW);
            bool a1 = (k1 <= qg) && (qg - k1 < WINDOW);
            v0 = a0 ? v0 : NEG_INF_F;
            v1 = a1 ? v1 : NEG_INF_F;
            float rm = fmaxf(v0, v1);
            rm = fmaxf(rm, __shfl_xor(rm, 1));
            rm = fmaxf(rm, __shfl_xor(rm, 2));
            rm = fmaxf(rm, __shfl_xor(rm, 4));
            rm = fmaxf(rm, __shfl_xor(rm, 8));
            float mn = fmaxf(m_run[r], rm);
            alpha[r] = __expf(m_run[r] - mn);
            p0v[r] = __expf(v0 - mn);
            p1v[r] = __expf(v1 - mn);
            float rs = p0v[r] + p1v[r];
            rs += __shfl_xor(rs, 1);
            rs += __shfl_xor(rs, 2);
            rs += __shfl_xor(rs, 4);
            rs += __shfl_xor(rs, 8);
            l_run[r] = l_run[r] * alpha[r] + rs;
            m_run[r] = mn;
        }
#pragma unroll
        for (int c = 0; c < 8; ++c)
#pragma unroll
            for (int r = 0; r < 8; ++r) acc[c][r] *= alpha[r];

        // ---- Transpose P through per-wave LDS buffer (C layout -> A layout) ----
        _Float16* pw = &Ps[wid * 16 * PS_STRIDE];
#pragma unroll
        for (int r = 0; r < 8; ++r) {
            pw[(r + hi * 8) * PS_STRIDE + ln]      = (_Float16)p0v[r];
            pw[(r + hi * 8) * PS_STRIDE + ln + 16] = (_Float16)p1v[r];
        }
        asm volatile("s_wait_dscnt 0" ::: "memory");  // same-wave DS RAW
        const _Float16* pp = &pw[ln * PS_STRIDE + hi * 8];
        v16h pf = ld8x2(pp, pp + 16);

        // ---- PV: 8 chunks of 16 output dims, contraction over 32 keys ----
#pragma unroll
        for (int c = 0; c < 8; ++c) {
            const _Float16* vp = &Vt[(c * 16 + ln) * VT_STRIDE + hi * 8];
            v16h vb = ld8x2(vp, vp + 16);
            acc[c] = wmma16(pf, vb, acc[c]);
        }
    }

    // ---- Epilogue: divide by denom, write out[b, s, h, d] ----
#pragma unroll
    for (int c = 0; c < 8; ++c) {
#pragma unroll
        for (int r = 0; r < 8; ++r) {
            int qg = qb + wid * 16 + hi * 8 + r;
            out[((size_t)qg * NHEAD + h) * DHEAD + c * 16 + ln] = acc[c][r] / l_run[r];
        }
    }
}

extern "C" void kernel_launch(void* const* d_in, const int* in_sizes, int n_in,
                              void* d_out, int out_size, void* d_ws, size_t ws_size,
                              hipStream_t stream) {
    (void)n_in; (void)d_ws; (void)ws_size; (void)out_size;
    const float* q         = (const float*)d_in[0];
    const float* k         = (const float*)d_in[1];
    const float* v         = (const float*)d_in[2];
    const int*   positions = (const int*)d_in[3];
    const float* sinks     = (const float*)d_in[4];
    float*       out       = (float*)d_out;

    const int S = in_sizes[0] / (NHEAD * DHEAD);   // 2048
    dim3 grid(S / 64, NHEAD);
    dim3 block(128);
    attn_swa_sink_kernel<<<grid, block, 0, stream>>>(q, k, v, positions, sinks, out, S);
}
